// ProsodyPredictor_40767829574405
// MI455X (gfx1250) — matrix operand order repaired and checked by reading, not compiled, not run on hardware
//
#include <hip/hip_runtime.h>

// ---------------------------------------------------------------------------
// ProsodyPredictor (autoregressive GRU + projection) for gfx1250 / MI455X.
// Persistent cooperative kernel, bf16 WMMA (v_wmma_f32_16x16x32_bf16),
// fp32 state & accumulation, 3 grid barriers per timestep.
// ---------------------------------------------------------------------------

typedef __attribute__((ext_vector_type(16))) __bf16        v16bf;
typedef __attribute__((ext_vector_type(8)))  float         v8f;
typedef __attribute__((ext_vector_type(4)))  unsigned int  u32x4;

union ABu { v16bf v; u32x4 q[2]; };
static_assert(sizeof(ABu) == 32, "ABu must be 32 bytes");

constexpr int N  = 64;
constexpr int T  = 1024;
constexpr int IT = 512;            // I_TEXT
constexpr int H  = 512;
constexpr int P  = 256;
constexpr int G3 = 3 * H;          // 1536
constexpr int KI = IT + P;         // 768 (concat [x, prev])

constexpr int NBLOCKS  = 32;
constexpr int TPB      = 256;
constexpr int NWAVES   = NBLOCKS * (TPB / 32);   // 256 waves
constexpr int NTHREADS = NBLOCKS * TPB;          // 8192 threads

// ---- workspace layout (bytes, all 256-aligned) ----
constexpr size_t OFF_WIHB = 0;                       // 1536*768*2  = 2359296
constexpr size_t OFF_WHHB = 2359296;                 // 1536*512*2  = 1572864
constexpr size_t OFF_WPB  = 3932160;                 // 256*512*2   = 262144
constexpr size_t OFF_GA   = 4194304;                 // 64*1536*4   = 393216
constexpr size_t OFF_GB   = 4587520;                 // 64*1536*4   = 393216
constexpr size_t OFF_HF   = 4980736;                 // 64*512*4    = 131072
constexpr size_t OFF_HBF  = 5111808;                 // 64*512*2    = 65536
constexpr size_t OFF_INPB = 5177344;                 // 64*768*2    = 98304
constexpr size_t OFF_BAR  = 5275648;                 // barrier words

__device__ __forceinline__ unsigned short f2bf(float f) {
  unsigned u = __builtin_bit_cast(unsigned, f);
  u += 0x7FFFu + ((u >> 16) & 1u);                   // round-to-nearest-even
  return (unsigned short)(u >> 16);
}

__device__ __forceinline__ float sigm(float x) { return 1.0f / (1.0f + __expf(-x)); }

__device__ __forceinline__ v8f wmma_bf16(const ABu& a, const ABu& b, v8f c) {
  // (neg_a, A, neg_b, B, c_mod, C, reuse_a, reuse_b)
  return __builtin_amdgcn_wmma_f32_16x16x32_bf16(false, a.v, false, b.v,
                                                 (short)0, c, false, false);
}

// Sense-free generation barrier across the persistent grid (device scope).
__device__ __forceinline__ void grid_barrier(unsigned* cnt, unsigned* gen, unsigned nb) {
  __syncthreads();
  if (threadIdx.x == 0) {
    unsigned g = *(volatile unsigned*)gen;           // read generation BEFORE arriving
    __threadfence();                                 // release my phase's stores
    unsigned a = atomicAdd(cnt, 1u);
    if (a == nb - 1u) {
      *(volatile unsigned*)cnt = 0u;
      __threadfence();
      atomicAdd(gen, 1u);                            // release the barrier
    } else {
      while (*(volatile unsigned*)gen == g) __builtin_amdgcn_s_sleep(2);
    }
    __threadfence();                                 // acquire others' stores
  }
  __syncthreads();
}

// ---------------------------------------------------------------------------
// Weight conversion fp32 -> bf16 (row-major [out][k], idempotent).
// ---------------------------------------------------------------------------
__global__ void prep_weights(const float* __restrict__ Wih, const float* __restrict__ Whh,
                             const float* __restrict__ Wp,
                             unsigned short* __restrict__ Wihb,
                             unsigned short* __restrict__ Whhb,
                             unsigned short* __restrict__ Wpb) {
  const int stride = gridDim.x * blockDim.x;
  const int i0 = blockIdx.x * blockDim.x + threadIdx.x;
  for (int i = i0; i < G3 * KI; i += stride) Wihb[i] = f2bf(Wih[i]);
  for (int i = i0; i < G3 * H;  i += stride) Whhb[i] = f2bf(Whh[i]);
  for (int i = i0; i < P  * H;  i += stride) Wpb[i]  = f2bf(Wp[i]);
}

// ---------------------------------------------------------------------------
// Per-call state init: h=0, prev=0, x_0 -> bf16, barrier re-armed.
// ---------------------------------------------------------------------------
__global__ void init_state(const float* __restrict__ X, float* __restrict__ hf,
                           unsigned short* __restrict__ hbf,
                           unsigned short* __restrict__ inpb,
                           unsigned* __restrict__ bar) {
  const int stride = gridDim.x * blockDim.x;
  const int i0 = blockIdx.x * blockDim.x + threadIdx.x;
  for (int i = i0; i < N * H; i += stride) { hf[i] = 0.0f; hbf[i] = 0; }
  for (int i = i0; i < N * KI; i += stride) {
    const int n = i / KI, k = i - n * KI;
    inpb[i] = (k < IT) ? f2bf(X[(size_t)n * T * IT + k]) : (unsigned short)0;
  }
  if (i0 == 0) { bar[0] = 0u; bar[1] = 0u; }
}

// ---------------------------------------------------------------------------
// Persistent recurrent kernel: all T steps, grid-synchronized.
// ---------------------------------------------------------------------------
__global__ void __launch_bounds__(TPB) prosody_persistent(
    const float* __restrict__ X,
    const float* __restrict__ bih, const float* __restrict__ bhh,
    const float* __restrict__ bp,
    const unsigned short* __restrict__ Wihb,
    const unsigned short* __restrict__ Whhb,
    const unsigned short* __restrict__ Wpb,
    float* __restrict__ gA, float* __restrict__ gB,
    float* __restrict__ hf, unsigned short* __restrict__ hbf,
    unsigned short* __restrict__ inpb,
    unsigned* __restrict__ bar,
    float* __restrict__ out) {

  const int lane  = threadIdx.x & 31;
  const int wave  = threadIdx.x >> 5;
  const int gwave = blockIdx.x * (TPB / 32) + wave;    // 0..255
  const int l15   = lane & 15;
  const int lhalf = lane >> 4;                         // lane[4]
  const int gtid  = blockIdx.x * TPB + threadIdx.x;

  for (int t = 0; t < T; ++t) {
    // ---------------- Phase A: gi & gh tiles (384 tiles of 16x16) ----------
    for (int tile = gwave; tile < (N / 16) * (G3 / 16); tile += NWAVES) {
      const int m0 = (tile & 3) * 16;
      const int c0 = (tile >> 2) * 16;

      // gi = [x,prev] @ W_ih^T over K=768
      const unsigned short* Ar = inpb + (m0 + l15) * KI;
      const unsigned short* Br = Wihb + (size_t)(c0 + l15) * KI;
      v8f gi = {};
      for (int k0 = 0; k0 < KI; k0 += 32) {
        ABu a, b;
        a.q[0] = *(const u32x4*)(Ar + k0 + lhalf * 8);        // K 0..7 / 8..15
        a.q[1] = *(const u32x4*)(Ar + k0 + 16 + lhalf * 8);   // K 16..23 / 24..31
        const unsigned short* wp2 = Br + k0 + lhalf * 8;      // K 0..7 / 8..15, +16
        b.q[0] = *(const u32x4*)(wp2);
        b.q[1] = *(const u32x4*)(wp2 + 16);
        gi = wmma_bf16(a, b, gi);
      }

      // gh = h @ W_hh^T over K=512
      const unsigned short* Hr  = hbf  + (m0 + l15) * H;
      const unsigned short* Br2 = Whhb + (size_t)(c0 + l15) * H;
      v8f gh = {};
      for (int k0 = 0; k0 < H; k0 += 32) {
        ABu a, b;
        a.q[0] = *(const u32x4*)(Hr + k0 + lhalf * 8);
        a.q[1] = *(const u32x4*)(Hr + k0 + 16 + lhalf * 8);
        const unsigned short* wp2 = Br2 + k0 + lhalf * 8;
        b.q[0] = *(const u32x4*)(wp2);
        b.q[1] = *(const u32x4*)(wp2 + 16);
        gh = wmma_bf16(a, b, gh);
      }

      // D layout: VGPR r, lanes 0-15 -> M=r, lanes 16-31 -> M=8+r; N=lane&15
      const int col = c0 + l15;
      const int mb  = m0 + lhalf * 8;
      #pragma unroll
      for (int r = 0; r < 8; ++r) {
        gA[(mb + r) * G3 + col] = gi[r];
        gB[(mb + r) * G3 + col] = gh[r];
      }
    }
    grid_barrier(bar, bar + 1, NBLOCKS);

    // ---------------- Phase B: GRU gates, update h (fp32 + bf16) -----------
    for (int idx = gtid; idx < N * H; idx += NTHREADS) {
      const int n = idx >> 9;                 // H == 512
      const int j = idx & (H - 1);
      const float* ga = gA + n * G3;
      const float* gb = gB + n * G3;
      const float r = sigm(ga[j] + bih[j] + gb[j] + bhh[j]);
      const float z = sigm(ga[j + H] + bih[j + H] + gb[j + H] + bhh[j + H]);
      const float g = tanhf(ga[j + 2 * H] + bih[j + 2 * H] +
                            r * (gb[j + 2 * H] + bhh[j + 2 * H]));
      const float hn = (1.0f - z) * g + z * hf[idx];
      hf[idx]  = hn;
      hbf[idx] = f2bf(hn);
    }
    grid_barrier(bar, bar + 1, NBLOCKS);

    // ------- Phase C: pred GEMM (waves 0..63) || convert x_{t+1} -----------
    if (gwave < (N / 16) * (P / 16)) {        // 64 tiles
      const int m0 = (gwave & 3) * 16;
      const int c0 = (gwave >> 2) * 16;
      const unsigned short* Ar = hbf + (m0 + l15) * H;
      const unsigned short* Br = Wpb + (size_t)(c0 + l15) * H;
      v8f acc = {};
      for (int k0 = 0; k0 < H; k0 += 32) {
        ABu a, b;
        a.q[0] = *(const u32x4*)(Ar + k0 + lhalf * 8);
        a.q[1] = *(const u32x4*)(Ar + k0 + 16 + lhalf * 8);
        const unsigned short* wp2 = Br + k0 + lhalf * 8;
        b.q[0] = *(const u32x4*)(wp2);
        b.q[1] = *(const u32x4*)(wp2 + 16);
        acc = wmma_bf16(a, b, acc);
      }
      const int col = c0 + l15;
      const float bpv = bp[col];
      const int mb = m0 + lhalf * 8;
      #pragma unroll
      for (int r = 0; r < 8; ++r) {
        const float v = acc[r] + bpv;
        const int row = mb + r;
        out[(size_t)row * T * P + (size_t)t * P + col] = v;   // out[n][t][p]
        inpb[row * KI + IT + col] = f2bf(v);                  // prev for t+1
      }
    } else if (t + 1 < T) {
      // 192 spare waves convert x_{t+1} fp32 -> bf16 into inp buffer
      for (int idx = (gwave - 64) * 32 + lane; idx < N * IT;
           idx += (NWAVES - 64) * 32) {
        const int n = idx >> 9;               // IT == 512
        const int k = idx & (IT - 1);
        inpb[n * KI + k] = f2bf(X[(size_t)n * T * IT + (size_t)(t + 1) * IT + k]);
      }
    }
    grid_barrier(bar, bar + 1, NBLOCKS);
  }
}

// ---------------------------------------------------------------------------
extern "C" void kernel_launch(void* const* d_in, const int* in_sizes, int n_in,
                              void* d_out, int out_size, void* d_ws, size_t ws_size,
                              hipStream_t stream) {
  const float* X   = (const float*)d_in[0];   // [N,T,IT]
  // d_in[1] = text_lengths (unused by the reference math)
  const float* Wih = (const float*)d_in[2];   // [3H, IT+P]
  const float* Whh = (const float*)d_in[3];   // [3H, H]
  const float* bih = (const float*)d_in[4];   // [3H]
  const float* bhh = (const float*)d_in[5];   // [3H]
  const float* Wp  = (const float*)d_in[6];   // [P, H]
  const float* bpp = (const float*)d_in[7];   // [P]

  char* w = (char*)d_ws;
  unsigned short* Wihb = (unsigned short*)(w + OFF_WIHB);
  unsigned short* Whhb = (unsigned short*)(w + OFF_WHHB);
  unsigned short* Wpb  = (unsigned short*)(w + OFF_WPB);
  float*          gA   = (float*)(w + OFF_GA);
  float*          gB   = (float*)(w + OFF_GB);
  float*          hf   = (float*)(w + OFF_HF);
  unsigned short* hbf  = (unsigned short*)(w + OFF_HBF);
  unsigned short* inpb = (unsigned short*)(w + OFF_INPB);
  unsigned*       bar  = (unsigned*)(w + OFF_BAR);

  hipLaunchKernelGGL(prep_weights, dim3(512), dim3(256), 0, stream,
                     Wih, Whh, Wp, Wihb, Whhb, Wpb);
  hipLaunchKernelGGL(init_state, dim3(128), dim3(256), 0, stream,
                     X, hf, hbf, inpb, bar);
  hipLaunchKernelGGL(prosody_persistent, dim3(NBLOCKS), dim3(TPB), 0, stream,
                     X, bih, bhh, bpp, Wihb, Whhb, Wpb,
                     gA, gB, hf, hbf, inpb, bar, (float*)d_out);
}
